// MyDeeperGCN_65197603553903
// MI455X (gfx1250) — matrix operand order repaired and checked by reading, not compiled, not run on hardware
//
#include <hip/hip_runtime.h>
#include <hip/hip_bf16.h>

typedef __attribute__((ext_vector_type(16))) _Float16 v16h;
typedef __attribute__((ext_vector_type(8)))  _Float16 v8h;
typedef __attribute__((ext_vector_type(8)))  float    v8f;

static constexpr int NN   = 5000;   // nodes
static constexpr int EE   = 10000;  // edges
static constexpr int FIN  = 16;
static constexpr int FE   = 4;
static constexpr int H    = 96;
static constexpr int EH   = 16;
static constexpr int HID3 = 32;
static constexpr int LL   = 10;
static constexpr int CC   = 3;
// Extended K for fused edge-MLP GEMM: (HID3+1 bias row) * H = 33*96 = 3168 -> 99 chunks of 32
static constexpr int KCH  = 99;                 // K chunks of 32
static constexpr int CHUNK_HALVES = 6 * 32 * 16;          // 3072 halves per chunk
static constexpr int W2SW_PER_LAYER = KCH * CHUNK_HALVES; // halves per layer (=304128)
static constexpr int HIDW = 34;                 // hid row stride (33 used, padded)
static constexpr int KSPLIT = 2;                // K-split waves per 16-edge tile

// ---------------------------------------------------------------- utilities
__global__ void fill_f32_kernel(float* p, float v, int n) {
    int i = blockIdx.x * blockDim.x + threadIdx.x;
    if (i < n) p[i] = v;
}

__global__ void sub_kernel(float* d, const float* s, int n) {
    int i = blockIdx.x * blockDim.x + threadIdx.x;
    if (i < n) d[i] -= s[i];
}

__global__ void tof16_kernel(const float* __restrict__ src, _Float16* __restrict__ dst, int n) {
    int i = blockIdx.x * blockDim.x + threadIdx.x;
    if (i < n) dst[i] = (_Float16)src[i];
}

// ---------------------------------------------------------------- ChebConv
__global__ void deg_kernel(const float* __restrict__ eattr, const int* __restrict__ ei,
                           float* __restrict__ deg) {
    int e = blockIdx.x * blockDim.x + threadIdx.x;
    if (e >= EE) return;
    int r = ei[e], c = ei[EE + e];
    if (eattr[e * FE] == 0.0f && r != c) atomicAdd(&deg[r], 1.0f);
}

__global__ void edgew_kernel(const float* __restrict__ eattr, const int* __restrict__ ei,
                             const float* __restrict__ deg, float* __restrict__ we) {
    int e = blockIdx.x * blockDim.x + threadIdx.x;
    if (e >= EE) return;
    int r = ei[e], c = ei[EE + e];
    float m  = (eattr[e * FE] == 0.0f && r != c) ? 1.0f : 0.0f;
    float dr = deg[r], dc = deg[c];
    float ir = dr > 0.0f ? rsqrtf(dr) : 0.0f;
    float ic = dc > 0.0f ? rsqrtf(dc) : 0.0f;
    we[e] = -ir * ic * m;
}

// dst[col] += scale * w_e * src[row]   (dst pre-zeroed)
__global__ void spmv16_kernel(const float* __restrict__ src, float* __restrict__ dst,
                              const float* __restrict__ we, const int* __restrict__ ei,
                              float scale) {
    int t = blockIdx.x * blockDim.x + threadIdx.x;
    if (t >= EE * FIN) return;
    int e = t >> 4, f = t & 15;
    float w = we[e] * scale;
    if (w == 0.0f) return;
    atomicAdd(&dst[ei[EE + e] * FIN + f], w * src[ei[e] * FIN + f]);
}

// out[n,o] += (bias?) + sum_f Tx[n,f] * W[f,o]
__global__ void cheb_accum_kernel(const float* __restrict__ Tx, const float* __restrict__ W,
                                  const float* __restrict__ bias, float* __restrict__ out) {
    int t = blockIdx.x * blockDim.x + threadIdx.x;
    if (t >= NN * H) return;
    int n = t / H, o = t % H;
    float s = bias ? bias[o] : 0.0f;
    const float* tp = Tx + n * FIN;
#pragma unroll
    for (int f = 0; f < FIN; ++f) s += tp[f] * W[f * H + o];
    out[t] += s;
}

// ---------------------------------------------------------------- edge encoder + MLP first layer
__global__ void ea_kernel(const float* __restrict__ eattr, const float* __restrict__ ew,
                          const float* __restrict__ eb, float* __restrict__ ea) {
    int e = blockIdx.x * blockDim.x + threadIdx.x;
    if (e >= EE) return;
    const float* ap = eattr + e * FE;
#pragma unroll
    for (int j = 0; j < EH; ++j) {
        float s = eb[j];
#pragma unroll
        for (int f = 0; f < FE; ++f) s += ap[f] * ew[f * EH + j];
        ea[e * EH + j] = s;
    }
}

// hid[l][e][k] = (ea @ w1[l] + b1[l])[k] for k<32 ; hid[..][32]=1 (bias row of extended GEMM)
__global__ void hid_kernel(const float* __restrict__ ea, const float* __restrict__ w1,
                           const float* __restrict__ b1, _Float16* __restrict__ hid) {
    int t = blockIdx.x * blockDim.x + threadIdx.x;
    if (t >= EE * LL) return;
    int e = t % EE, l = t / EE;
    _Float16* hp = hid + ((size_t)l * EE + e) * HIDW;
    const float* eap = ea + e * EH;
    const float* w1p = w1 + (size_t)l * EH * HID3;
    const float* b1p = b1 + l * HID3;
    for (int k = 0; k < HID3; ++k) {
        float s = b1p[k];
#pragma unroll
        for (int j = 0; j < EH; ++j) s += eap[j] * w1p[j * HID3 + k];
        hp[k] = (_Float16)s;
    }
    hp[32] = (_Float16)1.0f;
    hp[33] = (_Float16)0.0f;
}

// ---------------------------------------------------------------- W2 fragment pre-swizzle
// dst layout: [l][kc][nt][lane(32)][half(16)], one contiguous 32B fragment per lane.
// Fragment element j of lane L (ISA B 32x16 layout): B[(j<8 ? 8*(L>>4)+j : 16+8*(L>>4)+j-8)][nt*16 + (L&15)]
// where B-row (within chunk kc) r maps to k = kc/3, h = (kc%3)*32 + r, col o.
// Value = w2[l][k][h*96+o] for k<32, else b2[l][h*96+o].
__global__ void w2sw_kernel(const float* __restrict__ w2, const float* __restrict__ b2,
                            _Float16* __restrict__ dst) {
    int t = blockIdx.x * blockDim.x + threadIdx.x;
    const int tot = LL * W2SW_PER_LAYER;
    if (t >= tot) return;
    int j   = t & 15;
    int Ln  = (t >> 4) & 31;
    int nt  = (t >> 9) % 6;
    int kc  = (t / (512 * 6)) % KCH;
    int l   = t / (512 * 6 * KCH);
    int lhi = Ln >> 4, lmod = Ln & 15;
    int r = (j < 8) ? (lhi * 8 + j) : (16 + lhi * 8 + (j - 8));
    int k = kc / 3;
    int h = (kc % 3) * 32 + r;
    int o = nt * 16 + lmod;
    float val = (k < HID3) ? w2[((size_t)l * HID3 + k) * (H * H) + h * H + o]
                           : b2[(size_t)l * (H * H) + h * H + o];
    dst[t] = (_Float16)val;
}

// ---------------------------------------------------------------- per-layer node prep
__global__ void lnrelu_kernel(const float* __restrict__ h, const float* __restrict__ g,
                              const float* __restrict__ b, float* __restrict__ v32,
                              _Float16* __restrict__ v16) {
    int n = blockIdx.x * blockDim.x + threadIdx.x;
    if (n >= NN) return;
    const float* hp = h + (size_t)n * H;
    float mu = 0.0f;
    for (int o = 0; o < H; ++o) mu += hp[o];
    mu *= (1.0f / H);
    float var = 0.0f;
    for (int o = 0; o < H; ++o) { float d = hp[o] - mu; var += d * d; }
    var *= (1.0f / H);
    float rs = rsqrtf(var + 1e-5f);
    for (int o = 0; o < H; ++o) {
        float t = (hp[o] - mu) * rs * g[o] + b[o];
        t = fmaxf(t, 0.0f);
        v32[(size_t)n * H + o] = t;
        v16[(size_t)n * H + o] = (_Float16)t;
    }
}

// out[n,o] = cb[o] + (resid? resid[n,o]) + sum_h v[n,h]*root[h,o]
__global__ __launch_bounds__(96) void agginit_kernel(const float* __restrict__ v,
                                                     const float* __restrict__ root,
                                                     const float* __restrict__ cb,
                                                     const float* __restrict__ resid,
                                                     float* __restrict__ out) {
    __shared__ float sv[H];
    int n = blockIdx.x;
    int o = threadIdx.x;
    sv[o] = v[(size_t)n * H + o];
    __syncthreads();
    float s = cb[o] + (resid ? resid[(size_t)n * H + o] : 0.0f);
#pragma unroll 8
    for (int hh = 0; hh < H; ++hh) s += sv[hh] * root[hh * H + o];
    out[(size_t)n * H + o] = s;
}

// ---------------------------------------------------------------- fused edge-MLP x message GEMM
// KSPLIT waves per 16-edge x 96-output tile; wave kh handles k in [k0,k1) of the 33 extended
// hidden rows (K' = 33*96 = 3168 as 99 chunks of 32). A (16x32 f16) is built in registers:
// A[m][K] = hid[e,k] * v16[row[e], h0+K] (packed f16 muls). All 6 B fragments of a chunk are
// hoisted into registers first so the 12-load clause overlaps the 6 WMMAs via partial waits;
// an unconditional (clamped) global_prefetch per chunk pulls the kc+3 chunk toward the WGP
// cache. f32 accumulation; partial results scatter-added to agg[col[e], o] with global f32
// atomics (which makes the K-split free).
__global__ __launch_bounds__(256) void nnconv_gemm_kernel(
    const _Float16* __restrict__ vf16,   // [N,96]
    const _Float16* __restrict__ hid,    // [E,HIDW] this layer
    const _Float16* __restrict__ w2s,    // [KCH][6][32][16] halves, this layer
    const int* __restrict__ ei,          // [2,E]
    float* __restrict__ agg)             // [N,96]
{
    const int wid  = (blockIdx.x * (blockDim.x >> 5)) + (threadIdx.x >> 5);
    const int lane = threadIdx.x & 31;
    const int tile = wid / KSPLIT;
    const int kh   = wid % KSPLIT;
    if (tile * 16 >= EE) return;

    const int k0 = kh ? 17 : 0;
    const int k1 = kh ? 33 : 17;

    const int m   = lane & 15;
    const int lhi = lane >> 4;
    const int em  = tile * 16 + m;

    // cache this lane's A-source halves: lane<16 needs K={0..7,16..23}, lane>=16 the other runs
    const int srow = ei[em];
    const _Float16* xr = vf16 + (size_t)srow * H;
    v8h xlo[3], xhi[3];
#pragma unroll
    for (int hs = 0; hs < 3; ++hs) {
        int base = hs * 32 + (lhi ? 8 : 0);
        xlo[hs] = *(const v8h*)(xr + base);        // K run 0..7 (or 8..15)
        xhi[hs] = *(const v8h*)(xr + base + 16);   // K run 16..23 (or 24..31)
    }
    const _Float16* hrow = hid + (size_t)em * HIDW;

    v8f acc[6];
#pragma unroll
    for (int nt = 0; nt < 6; ++nt)
#pragma unroll
        for (int i = 0; i < 8; ++i) acc[nt][i] = 0.0f;

    for (int k = k0; k < k1; ++k) {
        _Float16 hv = hrow[k];
        v8h hsp;
#pragma unroll
        for (int i = 0; i < 8; ++i) hsp[i] = hv;
#pragma unroll
        for (int hs = 0; hs < 3; ++hs) {
            const int kc = k * 3 + hs;
            const _Float16* bbase = w2s + ((size_t)kc * 6 * 32 + lane) * 16;

            // issue the whole chunk's fragment loads as one clause
            v16h bf[6];
#pragma unroll
            for (int nt = 0; nt < 6; ++nt)
                bf[nt] = *(const v16h*)(bbase + (size_t)nt * 32 * 16);

            // prefetch a future chunk; clamp the (wave-uniform) index instead of branching
            // so no exec-mask manipulation lands in the hot loop. Lane-strided 192B covers
            // the 6KB chunk; locality 3 requests the near (WGP) scope.
            {
                int pk = kc + 3 < KCH ? kc + 3 : KCH - 1;
                __builtin_prefetch((const char*)(w2s + (size_t)pk * CHUNK_HALVES) +
                                       lane * 192, 0, 3);
            }

            // build A while loads are in flight
            v8h plo = xlo[hs] * hsp;   // packed f16 muls
            v8h phi = xhi[hs] * hsp;
            v16h a;
#pragma unroll
            for (int i = 0; i < 8; ++i) { a[i] = plo[i]; a[i + 8] = phi[i]; }

#pragma unroll
            for (int nt = 0; nt < 6; ++nt)
                acc[nt] = __builtin_amdgcn_wmma_f32_16x16x32_f16(
                    false, a, false, bf[nt], (short)0, acc[nt], false, false);
        }
    }

    // scatter: accumulator element r of lane L is edge (r + 8*lhi), output col nt*16 + (L&15)
    int cols[8];
#pragma unroll
    for (int r = 0; r < 8; ++r) cols[r] = ei[EE + tile * 16 + r + 8 * lhi];
    const int o0 = lane & 15;
#pragma unroll
    for (int nt = 0; nt < 6; ++nt)
#pragma unroll
        for (int r = 0; r < 8; ++r)
            atomicAdd(&agg[(size_t)cols[r] * H + nt * 16 + o0], acc[nt][r]);
}

// ---------------------------------------------------------------- head
__global__ void final_kernel(const float* __restrict__ h, const float* __restrict__ g,
                             const float* __restrict__ b, const float* __restrict__ lw,
                             const float* __restrict__ lb, float* __restrict__ out) {
    int n = blockIdx.x * blockDim.x + threadIdx.x;
    if (n >= NN) return;
    const float* hp = h + (size_t)n * H;
    float mu = 0.0f;
    for (int o = 0; o < H; ++o) mu += hp[o];
    mu *= (1.0f / H);
    float var = 0.0f;
    for (int o = 0; o < H; ++o) { float d = hp[o] - mu; var += d * d; }
    var *= (1.0f / H);
    float rs = rsqrtf(var + 1e-5f);
    float a0 = lb[0], a1 = lb[1], a2 = lb[2];
    for (int o = 0; o < H; ++o) {
        float t = (hp[o] - mu) * rs * g[o] + b[o];
        t = fmaxf(t, 0.0f);
        a0 += t * lw[o * CC + 0];
        a1 += t * lw[o * CC + 1];
        a2 += t * lw[o * CC + 2];
    }
    out[n * CC + 0] = a0;
    out[n * CC + 1] = a1;
    out[n * CC + 2] = a2;
}

// ================================================================ host driver
extern "C" void kernel_launch(void* const* d_in, const int* in_sizes, int n_in,
                              void* d_out, int out_size, void* d_ws, size_t ws_size,
                              hipStream_t stream) {
    (void)in_sizes; (void)n_in; (void)out_size; (void)ws_size;
    const float* x     = (const float*)d_in[0];
    const int*   ei    = (const int*)d_in[1];
    const float* eattr = (const float*)d_in[2];
    const float* chebw = (const float*)d_in[3];
    const float* chebb = (const float*)d_in[4];
    const float* encw  = (const float*)d_in[5];
    const float* encb  = (const float*)d_in[6];
    const float* w1    = (const float*)d_in[7];
    const float* b1    = (const float*)d_in[8];
    const float* w2    = (const float*)d_in[9];
    const float* b2    = (const float*)d_in[10];
    const float* root  = (const float*)d_in[11];
    const float* cb    = (const float*)d_in[12];
    const float* lng   = (const float*)d_in[13];
    const float* lnb   = (const float*)d_in[14];
    const float* lw    = (const float*)d_in[15];
    const float* lb    = (const float*)d_in[16];
    float* out = (float*)d_out;

    size_t off = 0;
    auto carve = [&](size_t bytes) -> void* {
        off = (off + 255) & ~(size_t)255;
        void* p = (char*)d_ws + off;
        off += bytes;
        return p;
    };
    float*     deg   = (float*)carve((size_t)NN * 4);
    float*     T1    = (float*)carve((size_t)NN * FIN * 4);
    float*     T2    = (float*)carve((size_t)NN * FIN * 4);
    float*     T3    = (float*)carve((size_t)NN * FIN * 4);
    float*     we    = (float*)carve((size_t)EE * 4);
    float*     hA    = (float*)carve((size_t)NN * H * 4);
    float*     hB    = (float*)carve((size_t)NN * H * 4);
    float*     v32   = (float*)carve((size_t)NN * H * 4);
    float*     eabuf = (float*)carve((size_t)EE * EH * 4);
    _Float16*  v16p  = (_Float16*)carve((size_t)NN * H * 2);
    _Float16*  hid   = (_Float16*)carve((size_t)LL * EE * HIDW * 2);
    _Float16*  w2s   = (_Float16*)carve((size_t)LL * W2SW_PER_LAYER * 2);

    const int TPB = 256;
    auto nb = [](int n, int t) { return (n + t - 1) / t; };

    // ---- ChebConv node encoder
    fill_f32_kernel<<<nb(NN, TPB), TPB, 0, stream>>>(deg, 0.0f, NN);
    deg_kernel<<<nb(EE, TPB), TPB, 0, stream>>>(eattr, ei, deg);
    edgew_kernel<<<nb(EE, TPB), TPB, 0, stream>>>(eattr, ei, deg, we);

    fill_f32_kernel<<<nb(NN * H, TPB), TPB, 0, stream>>>(hA, 0.0f, NN * H);
    cheb_accum_kernel<<<nb(NN * H, TPB), TPB, 0, stream>>>(x, chebw + 0 * FIN * H, chebb, hA);
    // Tx1 = L x
    fill_f32_kernel<<<nb(NN * FIN, TPB), TPB, 0, stream>>>(T1, 0.0f, NN * FIN);
    spmv16_kernel<<<nb(EE * FIN, TPB), TPB, 0, stream>>>(x, T1, we, ei, 1.0f);
    cheb_accum_kernel<<<nb(NN * H, TPB), TPB, 0, stream>>>(T1, chebw + 1 * FIN * H, (const float*)nullptr, hA);
    // Tx2 = 2 L Tx1 - x
    fill_f32_kernel<<<nb(NN * FIN, TPB), TPB, 0, stream>>>(T2, 0.0f, NN * FIN);
    spmv16_kernel<<<nb(EE * FIN, TPB), TPB, 0, stream>>>(T1, T2, we, ei, 2.0f);
    sub_kernel<<<nb(NN * FIN, TPB), TPB, 0, stream>>>(T2, x, NN * FIN);
    cheb_accum_kernel<<<nb(NN * H, TPB), TPB, 0, stream>>>(T2, chebw + 2 * FIN * H, (const float*)nullptr, hA);
    // Tx3 = 2 L Tx2 - Tx1
    fill_f32_kernel<<<nb(NN * FIN, TPB), TPB, 0, stream>>>(T3, 0.0f, NN * FIN);
    spmv16_kernel<<<nb(EE * FIN, TPB), TPB, 0, stream>>>(T2, T3, we, ei, 2.0f);
    sub_kernel<<<nb(NN * FIN, TPB), TPB, 0, stream>>>(T3, T1, NN * FIN);
    cheb_accum_kernel<<<nb(NN * H, TPB), TPB, 0, stream>>>(T3, chebw + 3 * FIN * H, (const float*)nullptr, hA);
    // Tx4 = 2 L Tx3 - Tx2   (reuse T1)
    fill_f32_kernel<<<nb(NN * FIN, TPB), TPB, 0, stream>>>(T1, 0.0f, NN * FIN);
    spmv16_kernel<<<nb(EE * FIN, TPB), TPB, 0, stream>>>(T3, T1, we, ei, 2.0f);
    sub_kernel<<<nb(NN * FIN, TPB), TPB, 0, stream>>>(T1, T2, NN * FIN);
    cheb_accum_kernel<<<nb(NN * H, TPB), TPB, 0, stream>>>(T1, chebw + 4 * FIN * H, (const float*)nullptr, hA);

    // ---- edge encoder, per-layer hidden, W2 fragment swizzle
    ea_kernel<<<nb(EE, TPB), TPB, 0, stream>>>(eattr, encw, encb, eabuf);
    hid_kernel<<<nb(EE * LL, TPB), TPB, 0, stream>>>(eabuf, w1, b1, hid);
    w2sw_kernel<<<nb(LL * W2SW_PER_LAYER, TPB), TPB, 0, stream>>>(w2, b2, w2s);

    // ---- NNConv layers
    float* hcur  = hA;
    float* hnext = hB;
    const int total_waves = (EE / 16) * KSPLIT;      // 1250
    const int gemm_blocks = (total_waves + 7) / 8;   // 8 waves per 256-thread block
    for (int i = 0; i < LL; ++i) {
        const float* vsrc;
        if (i == 0) {
            tof16_kernel<<<nb(NN * H, TPB), TPB, 0, stream>>>(hcur, v16p, NN * H);
            vsrc = hcur;
        } else {
            lnrelu_kernel<<<nb(NN, 128), 128, 0, stream>>>(hcur, lng + i * H, lnb + i * H, v32, v16p);
            vsrc = v32;
        }
        agginit_kernel<<<NN, H, 0, stream>>>(vsrc, root + (size_t)i * H * H, cb + i * H,
                                             (i ? hcur : (const float*)nullptr), hnext);
        nnconv_gemm_kernel<<<gemm_blocks, 256, 0, stream>>>(
            v16p, hid + (size_t)i * EE * HIDW, w2s + (size_t)i * W2SW_PER_LAYER, ei, hnext);
        float* t = hcur; hcur = hnext; hnext = t;
    }

    // ---- head (uses layer-0 LN params)
    final_kernel<<<nb(NN, 128), 128, 0, stream>>>(hcur, lng, lnb, lw, lb, out);
}
